// Generator_44676249813609
// MI455X (gfx1250) — matrix operand order repaired and checked by reading, not compiled, not run on hardware
//
#include <hip/hip_runtime.h>
#include <hip/hip_bf16.h>
#include <stdint.h>

// Problem constants (match reference)
#define BB 32
#define SS 256
#define VV 10000
#define HH 1024
#define EE 512
#define GG 4096   // 4*H gate width
#define NEGF (-1e30f)

typedef __bf16 bf16;
typedef __attribute__((ext_vector_type(16))) __bf16 v16bf;
typedef __attribute__((ext_vector_type(8)))  float  v8f;
typedef int v4i_ __attribute__((vector_size(16)));   // 16B, 16B-aligned

#ifndef __has_builtin
#define __has_builtin(x) 0
#endif

#if defined(__HIP_DEVICE_COMPILE__) && __has_builtin(__builtin_amdgcn_global_load_async_to_lds_b128)
#define HAVE_ASYNC_LDS 1
#else
#define HAVE_ASYNC_LDS 0
#endif

union BFrag { v4i_ q[2]; v16bf v; };

// ---------------------------------------------------------------------------
// WMMA fragment loaders per CDNA5 ISA 7.12.2 (wave32 layouts)
// ---------------------------------------------------------------------------

// A-matrix 16x32 bf16 fragment from row-major A[M][K] (lda elements).
// lanes 0-15: M=lane, V0..V3 -> K pairs {0..7}, V4..V7 -> {16..23}
// lanes16-31: M=lane-16, V0..V3 -> {8..15},     V4..V7 -> {24..31}
// => two contiguous 16B runs at element offsets (k0+8*hi) and (k0+8*hi+16).
__device__ __forceinline__ v16bf load_a_frag(const bf16* A, int lda, int m0, int k0, int lane) {
  BFrag r;
  const int hi = lane >> 4;
  const int m  = m0 + (lane & 15);
  const v4i_* p = (const v4i_*)(A + (size_t)m * lda + k0 + (hi << 3));
  r.q[0] = p[0];          // 16 bytes: K pairs base..base+7
  r.q[1] = p[2];          // +32 bytes: K pairs base+16..base+23
  return r.v;
}

// B-matrix 32x16 bf16 fragment from K-major weight Wt[K][N] (ldb = N elements).
// lane l holds column N = n0+(l&15); elements s<8 -> K = k0+8*hi+s, s>=8 -> K = k0+16+8*hi+(s-8).
__device__ __forceinline__ v16bf load_b_frag(const bf16* Wt, int ldb, int k0, int n0, int lane) {
  BFrag r;
  const int hi = lane >> 4;
  const bf16* p = Wt + (size_t)(k0 + (hi << 3)) * ldb + n0 + (lane & 15);
  #pragma unroll
  for (int s = 0; s < 8; ++s) {
    r.v[s]     = p[(size_t)s * ldb];
    r.v[8 + s] = p[(size_t)(16 + s) * ldb];
  }
  return r.v;
}

__device__ __forceinline__ v8f wmma_bf16(v16bf a, v16bf b, v8f c) {
  return __builtin_amdgcn_wmma_f32_16x16x32_bf16(false, a, false, b, (short)0, c, false, false);
}

__device__ __forceinline__ void wait_async0() {
#if __has_builtin(__builtin_amdgcn_s_wait_asynccnt)
  __builtin_amdgcn_s_wait_asynccnt(0);
#else
  asm volatile("s_wait_asynccnt 0" ::: "memory");
#endif
}

// Stage 32 rows of A (32 x K bf16, row-major) into LDS panel As (pitch KT),
// starting at column offset kofs. 256-thread cooperative; 16B chunks via
// the CDNA5 async global->LDS path (ASYNCcnt), fallback to plain copies.
__device__ __forceinline__ void stage_rows(bf16* As, int KT, const bf16* A, int K, int kofs) {
  const int tid = threadIdx.x;
#if HAVE_ASYNC_LDS
  const int rowCh = K >> 3;                 // 16-byte chunks per row
  for (int c = tid; c < 32 * rowCh; c += 256) {
    int m = c / rowCh, o = c - m * rowCh;
    const void* g = (const void*)(A + (size_t)m * K + (size_t)o * 8);
    void*       l = (void*)(As + (size_t)m * KT + kofs + (size_t)o * 8);
    __builtin_amdgcn_global_load_async_to_lds_b128(
        (__attribute__((address_space(1))) v4i_*)(uintptr_t)g,
        (__attribute__((address_space(3))) v4i_*)(uint32_t)(uintptr_t)l,
        0, 0);
  }
#else
  const int rowDw = K >> 1;                 // dwords per row
  const uint32_t* src = (const uint32_t*)A;
  for (int c = tid; c < 32 * rowDw; c += 256) {
    int m = c / rowDw, o = c - m * rowDw;
    *(uint32_t*)(As + (size_t)m * KT + kofs + 2 * o) = src[c];
  }
#endif
}

// ---------------------------------------------------------------------------
// M=32 GEMM core: C(32xN,f32) = A1@W1t + [A2@W2t] + bias1 + bias2
// A panel staged in LDS once per block (8 waves share); one wave per 16-col strip.
// ---------------------------------------------------------------------------
__device__ __forceinline__ void gemm32_core(bf16* As,
                                            const bf16* A1, int K1,
                                            const bf16* A2, int K2,
                                            const bf16* W1t, const bf16* W2t,
                                            const float* bias1, const float* bias2,
                                            float* C, int N, int blockId) {
  const int KT = K1 + K2;
  stage_rows(As, KT, A1, K1, 0);
  if (A2) stage_rows(As, KT, A2, K2, K1);
#if HAVE_ASYNC_LDS
  wait_async0();
#endif
  __syncthreads();

  const int lane = threadIdx.x & 31;
  const int wave = blockId * 8 + (threadIdx.x >> 5);
  const int n0 = wave << 4;
  if (n0 >= N) return;                      // wave-uniform: EXEC stays all-ones
  v8f acc0 = {};
  v8f acc1 = {};
  for (int k0 = 0; k0 < KT; k0 += 32) {
    const bf16* W = (k0 < K1) ? W1t : W2t;  // wave-uniform select
    const int kk  = (k0 < K1) ? k0 : k0 - K1;
    v16bf bfr = load_b_frag(W, N, kk, n0, lane);
    v16bf a0  = load_a_frag(As, KT, 0,  k0, lane);   // ds_load_b128 path
    v16bf a1  = load_a_frag(As, KT, 16, k0, lane);
    acc0 = wmma_bf16(a0, bfr, acc0);
    acc1 = wmma_bf16(a1, bfr, acc1);
  }
  const int n  = n0 + (lane & 15);
  const int hi = lane >> 4;
  float bv = 0.f;
  if (bias1) bv += bias1[n];
  if (bias2) bv += bias2[n];
  #pragma unroll
  for (int r = 0; r < 8; ++r) {
    C[(size_t)(r + 8 * hi) * N + n]      = acc0[r] + bv;   // rows 0..15
    C[(size_t)(16 + r + 8 * hi) * N + n] = acc1[r] + bv;   // rows 16..31
  }
}

__global__ void gemm32_kernel(const bf16* __restrict__ A1, int K1,
                              const bf16* __restrict__ A2, int K2,
                              const bf16* __restrict__ W1t,
                              const bf16* __restrict__ W2t,
                              const float* __restrict__ bias1,
                              const float* __restrict__ bias2,
                              float* __restrict__ C, int N) {
  extern __shared__ bf16 As[];
  gemm32_core(As, A1, K1, A2, K2, W1t, W2t, bias1, bias2, C, N, blockIdx.x);
}

// Paired launch: first half of grid computes Cq = Aq@Wqt, second half Ck = Ak@Wkt
// (both M=32, K=H, N=H, no bias). Saves one launch on the serial chain.
__global__ void gemm32_qk_kernel(const bf16* __restrict__ Aq, const bf16* __restrict__ Wqt,
                                 float* __restrict__ Cq,
                                 const bf16* __restrict__ Ak, const bf16* __restrict__ Wkt,
                                 float* __restrict__ Ck) {
  extern __shared__ bf16 As[];
  const int half = gridDim.x >> 1;
  const bool second = (int)blockIdx.x >= half;
  const int blk = second ? (int)blockIdx.x - half : (int)blockIdx.x;
  gemm32_core(As, second ? Ak : Aq, HH, (const bf16*)nullptr, 0,
              second ? Wkt : Wqt, (const bf16*)nullptr,
              (const float*)nullptr, (const float*)nullptr,
              second ? Ck : Cq, HH, blk);
}

// ---------------------------------------------------------------------------
// Init / staging kernels
// ---------------------------------------------------------------------------
__global__ void zero_kernel(float* p, int n) {
  int i = blockIdx.x * blockDim.x + threadIdx.x;
  if (i < n) p[i] = 0.f;
}

__global__ void cvt_kernel(const float* __restrict__ src, bf16* __restrict__ dst, int n) {
  int i = blockIdx.x * blockDim.x + threadIdx.x;
  if (i < n) dst[i] = (bf16)src[i];
}

// src[N][K] (row-major) -> dst[K][N]  (K-major for WMMA B-fragments)
__global__ void cvtT_kernel(const float* __restrict__ src, bf16* __restrict__ dst,
                            int Nrows, int Kcols) {
  int i = blockIdx.x * blockDim.x + threadIdx.x;
  if (i < Nrows * Kcols) {
    int nn = i / Kcols, kk = i - nn * Kcols;
    dst[(size_t)kk * Nrows + nn] = (bf16)src[i];
  }
}

// embedding gather (bf16) + bf16 casts of current h_e, h_d
__global__ void stage_kernel(const int* __restrict__ x, int t,
                             const bf16* __restrict__ emb_bf, bf16* __restrict__ e_bf,
                             const float* __restrict__ h_e, bf16* __restrict__ h_e_bf,
                             const float* __restrict__ h_d, bf16* __restrict__ h_d_bf) {
  int idx = blockIdx.x * blockDim.x + threadIdx.x;
  if (idx < BB * EE) {
    int b = idx / EE, k = idx - b * EE;
    int tok = x[b * SS + t];
    e_bf[idx] = emb_bf[(size_t)tok * EE + k];
  } else if (idx < BB * EE + BB * HH) {
    int i = idx - BB * EE;
    h_e_bf[i] = (bf16)h_e[i];
  } else if (idx < BB * EE + 2 * BB * HH) {
    int i = idx - BB * EE - BB * HH;
    h_d_bf[i] = (bf16)h_d[i];
  }
}

__device__ __forceinline__ float sigf(float x) { return 1.f / (1.f + __expf(-x)); }

// LSTM pointwise: gates g[32][4096] (i,f,g,o) -> new h,c (+ bf16 h, + optional ann[t])
__global__ void lstm_update_kernel(const float* __restrict__ g,
                                   float* __restrict__ h, float* __restrict__ c,
                                   bf16* __restrict__ h_bf, float* __restrict__ ann_row) {
  int idx = blockIdx.x * blockDim.x + threadIdx.x;
  if (idx >= BB * HH) return;
  int b = idx >> 10, j = idx & (HH - 1);
  const float* gb = g + (size_t)b * GG;
  float ig = sigf(gb[j]);
  float fg = sigf(gb[HH + j]);
  float gg = tanhf(gb[2 * HH + j]);
  float og = sigf(gb[3 * HH + j]);
  float cn = fg * c[idx] + ig * gg;
  float hn = og * tanhf(cn);
  c[idx] = cn;
  h[idx] = hn;
  h_bf[idx] = (bf16)hn;
  if (ann_row) ann_row[idx] = hn;   // ann[t][b][j], idx == b*H+j
}

// scores[s][b] = sum_h tanh(q[b][h] + keys[s][b][h] + b_att[h]) * v_att[h]  (s<t), else NEG
__global__ void attn_scores_kernel(const float* __restrict__ q,
                                   const float* __restrict__ keys,
                                   const float* __restrict__ b_att,
                                   const float* __restrict__ v_att,
                                   float* __restrict__ scores, int t) {
  int lane = threadIdx.x & 31;
  int p = blockIdx.x * (blockDim.x >> 5) + (threadIdx.x >> 5);
  if (p >= SS * BB) return;
  int s = p >> 5, b = p & 31;
  float acc;
  if (s < t) {
    const float* kk = keys + (size_t)(s * BB + b) * HH;
    const float* qq = q + (size_t)b * HH;
    acc = 0.f;
    for (int hh = lane; hh < HH; hh += 32)
      acc += tanhf(qq[hh] + kk[hh] + b_att[hh]) * v_att[hh];
    #pragma unroll
    for (int off = 16; off; off >>= 1) acc += __shfl_xor(acc, off, 32);
  } else {
    acc = NEGF;
  }
  if (lane == 0) scores[s * BB + b] = acc;
}

// Fused: softmax over s (length S==blockDim) + ctx[b][:] = sum_s alpha[s]*ann[s][b][:]
// One block per batch element b.
__global__ void attn_softmax_ctx_kernel(const float* __restrict__ scores,
                                        const float* __restrict__ ann,
                                        float* __restrict__ ctx, bf16* __restrict__ ctx_bf,
                                        int t) {
  __shared__ float sAl[SS];
  __shared__ float red[256];
  const int b = blockIdx.x, tid = threadIdx.x;   // tid == s
  float sc = scores[tid * BB + b];
  red[tid] = sc; __syncthreads();
  for (int o = 128; o; o >>= 1) { if (tid < o) red[tid] = fmaxf(red[tid], red[tid + o]); __syncthreads(); }
  float m = red[0]; __syncthreads();
  float e = __expf(sc - m);
  red[tid] = e; __syncthreads();
  for (int o = 128; o; o >>= 1) { if (tid < o) red[tid] += red[tid + o]; __syncthreads(); }
  sAl[tid] = e / red[0];
  __syncthreads();
  for (int h = tid; h < HH; h += 256) {
    float acc = 0.f;
    for (int s = 0; s < t; ++s)                     // rows >= t are zero / alpha ~ 0
      acc += sAl[s] * ann[(size_t)(s * BB + b) * HH + h];
    ctx[b * HH + h] = acc;
    ctx_bf[b * HH + h] = (bf16)acc;
  }
}

// Fused: log_softmax over V for each b, write out[b][t][:]. One block per b.
__global__ void logits_finalize_kernel(const float* __restrict__ logits,
                                       float* __restrict__ out, int t) {
  __shared__ float red[256];
  const int b = blockIdx.x, tid = threadIdx.x;
  const float* row = logits + (size_t)b * VV;
  float m = -3.4e38f;
  for (int v = tid; v < VV; v += 256) m = fmaxf(m, row[v]);
  red[tid] = m; __syncthreads();
  for (int o = 128; o; o >>= 1) { if (tid < o) red[tid] = fmaxf(red[tid], red[tid + o]); __syncthreads(); }
  m = red[0]; __syncthreads();
  float sum = 0.f;
  for (int v = tid; v < VV; v += 256) sum += __expf(row[v] - m);
  red[tid] = sum; __syncthreads();
  for (int o = 128; o; o >>= 1) { if (tid < o) red[tid] += red[tid + o]; __syncthreads(); }
  const float lse = m + logf(red[0]);
  float* orow = out + (size_t)b * SS * VV + (size_t)t * VV;
  for (int v = tid; v < VV; v += 256) orow[v] = row[v] - lse;
}

// ---------------------------------------------------------------------------
// Host driver
// ---------------------------------------------------------------------------
extern "C" void kernel_launch(void* const* d_in, const int* in_sizes, int n_in,
                              void* d_out, int out_size, void* d_ws, size_t ws_size,
                              hipStream_t stream) {
  (void)in_sizes; (void)n_in; (void)out_size; (void)ws_size;
  const int*   x        = (const int*)  d_in[0];
  const float* emb      = (const float*)d_in[1];
  const float* W_ih_enc = (const float*)d_in[2];
  const float* W_hh_enc = (const float*)d_in[3];
  const float* b_ih_enc = (const float*)d_in[4];
  const float* b_hh_enc = (const float*)d_in[5];
  const float* W_ih_dec = (const float*)d_in[6];
  const float* W_hh_dec = (const float*)d_in[7];
  const float* b_ih_dec = (const float*)d_in[8];
  const float* b_hh_dec = (const float*)d_in[9];
  const float* W_out    = (const float*)d_in[10];
  const float* b_out    = (const float*)d_in[11];
  const float* Wq       = (const float*)d_in[12];
  const float* Wk       = (const float*)d_in[13];
  const float* b_att    = (const float*)d_in[14];
  const float* v_att    = (const float*)d_in[15];
  float* out = (float*)d_out;

  // Workspace carve-out (256B aligned)
  char* w = (char*)d_ws;
  auto alloc = [&](size_t bytes) { char* p = w; w += (bytes + 255) & ~(size_t)255; return p; };
  bf16* emb_bf    = (bf16*)alloc((size_t)VV * EE * 2);
  bf16* Wt_ih_enc = (bf16*)alloc((size_t)EE * GG * 2);
  bf16* Wt_hh_enc = (bf16*)alloc((size_t)HH * GG * 2);
  bf16* Wt_ih_dec = (bf16*)alloc((size_t)HH * GG * 2);
  bf16* Wt_hh_dec = (bf16*)alloc((size_t)HH * GG * 2);
  bf16* Wt_out    = (bf16*)alloc((size_t)HH * VV * 2);
  bf16* Wq_bf     = (bf16*)alloc((size_t)HH * HH * 2);   // already K-major (h @ Wq)
  bf16* Wk_bf     = (bf16*)alloc((size_t)HH * HH * 2);
  // zeroed region (contiguous): h_e, c_e, h_d, c_d, ann, keys
  float* h_e  = (float*)alloc((size_t)BB * HH * 4);
  float* c_e  = (float*)alloc((size_t)BB * HH * 4);
  float* h_d  = (float*)alloc((size_t)BB * HH * 4);
  float* c_d  = (float*)alloc((size_t)BB * HH * 4);
  float* ann  = (float*)alloc((size_t)SS * BB * HH * 4);
  float* keys = (float*)alloc((size_t)SS * BB * HH * 4);
  // transient buffers (fully rewritten each step)
  bf16*  e_bf   = (bf16*)alloc((size_t)BB * EE * 2);
  bf16*  h_e_bf = (bf16*)alloc((size_t)BB * HH * 2);
  bf16*  h_d_bf = (bf16*)alloc((size_t)BB * HH * 2);
  bf16*  ctx_bf = (bf16*)alloc((size_t)BB * HH * 2);
  float* gates  = (float*)alloc((size_t)BB * GG * 4);
  float* q      = (float*)alloc((size_t)BB * HH * 4);
  float* ctx    = (float*)alloc((size_t)BB * HH * 4);
  float* scores = (float*)alloc((size_t)SS * BB * 4);
  float* logits = (float*)alloc((size_t)BB * VV * 4);

  const dim3 blk(256);
  auto nb = [](int n) { return dim3((n + 255) / 256); };

  // --- one-time (per launch) conversions: fp32 -> bf16, transposed to K-major
  cvt_kernel <<<nb(VV * EE), blk, 0, stream>>>(emb, emb_bf, VV * EE);
  cvtT_kernel<<<nb(GG * EE), blk, 0, stream>>>(W_ih_enc, Wt_ih_enc, GG, EE);
  cvtT_kernel<<<nb(GG * HH), blk, 0, stream>>>(W_hh_enc, Wt_hh_enc, GG, HH);
  cvtT_kernel<<<nb(GG * HH), blk, 0, stream>>>(W_ih_dec, Wt_ih_dec, GG, HH);
  cvtT_kernel<<<nb(GG * HH), blk, 0, stream>>>(W_hh_dec, Wt_hh_dec, GG, HH);
  cvtT_kernel<<<nb(VV * HH), blk, 0, stream>>>(W_out, Wt_out, VV, HH);
  cvt_kernel <<<nb(HH * HH), blk, 0, stream>>>(Wq, Wq_bf, HH * HH);
  cvt_kernel <<<nb(HH * HH), blk, 0, stream>>>(Wk, Wk_bf, HH * HH);

  // --- zero states + ann + keys (contiguous)
  const int nzero = 4 * BB * HH + 2 * SS * BB * HH;
  zero_kernel<<<nb(nzero), blk, 0, stream>>>(h_e, nzero);

  // LDS panel sizes (32 x KT bf16)
  const size_t shEnc = (size_t)32 * (EE + HH) * 2;   //  96 KB
  const size_t shDec = (size_t)32 * (HH + HH) * 2;   // 128 KB
  const size_t shH   = (size_t)32 * HH * 2;          //  64 KB

  const dim3 gGates(GG / 16 / 8);        // 32 blocks (256 waves)
  const dim3 gQK(2 * HH / 16 / 8);       // 16 blocks (q + keys)
  const dim3 gV((VV / 16 + 7) / 8);      // 79 blocks (625 waves)
  const dim3 gPairs((SS * BB) / 8);      // 1024 blocks
  const dim3 gB(BB);                     // 32 blocks

  for (int t = 0; t < SS; ++t) {
    // stage: e = emb[x[:,t]] (bf16), cast h_e/h_d to bf16
    stage_kernel<<<nb(BB * EE + 2 * BB * HH), blk, 0, stream>>>(
        x, t, emb_bf, e_bf, h_e, h_e_bf, h_d, h_d_bf);

    // encoder gates: e@W_ih^T + h_e@W_hh^T + b_ih + b_hh
    gemm32_kernel<<<gGates, blk, shEnc, stream>>>(e_bf, EE, h_e_bf, HH,
                                                  Wt_ih_enc, Wt_hh_enc,
                                                  b_ih_enc, b_hh_enc, gates, GG);
    // encoder cell update; append ann[t] = new h_e; emit new h_e_bf
    lstm_update_kernel<<<nb(BB * HH), blk, 0, stream>>>(
        gates, h_e, c_e, h_e_bf, ann + (size_t)t * BB * HH);

    // q = h_d(old) @ Wq ; keys[t] = h_e(new) @ Wk  (single paired launch)
    gemm32_qk_kernel<<<gQK, blk, shH, stream>>>(h_d_bf, Wq_bf, q,
                                                h_e_bf, Wk_bf,
                                                keys + (size_t)t * BB * HH);

    // additive attention over s < t
    attn_scores_kernel     <<<gPairs, blk, 0, stream>>>(q, keys, b_att, v_att, scores, t);
    attn_softmax_ctx_kernel<<<gB, blk, 0, stream>>>(scores, ann, ctx, ctx_bf, t);

    // decoder gates: ctx@W_ih^T + h_d(old)@W_hh^T + biases; then cell update
    gemm32_kernel<<<gGates, blk, shDec, stream>>>(ctx_bf, HH, h_d_bf, HH,
                                                  Wt_ih_dec, Wt_hh_dec,
                                                  b_ih_dec, b_hh_dec, gates, GG);
    lstm_update_kernel<<<nb(BB * HH), blk, 0, stream>>>(
        gates, h_d, c_d, h_d_bf, (float*)nullptr);

    // vocab projection + fused log_softmax + writeout
    gemm32_kernel<<<gV, blk, shH, stream>>>(h_d_bf, HH, (const bf16*)nullptr, 0,
                                            Wt_out, (const bf16*)nullptr,
                                            b_out, (const float*)nullptr, logits, VV);
    logits_finalize_kernel<<<gB, blk, 0, stream>>>(logits, out, t);
  }
}
